// CoAttention_74912819577424
// MI455X (gfx1250) — matrix operand order repaired and checked by reading, not compiled, not run on hardware
//
#include <hip/hip_runtime.h>

#define B_   8
#define SQ_  1024
#define SKV_ 1024
#define H_   1024
#define NH_  16
#define HD_  64

typedef float          v8f   __attribute__((ext_vector_type(8)));
typedef float          f4    __attribute__((ext_vector_type(4)));
typedef __bf16         v16bf __attribute__((ext_vector_type(16)));
typedef unsigned int   u32x4 __attribute__((ext_vector_type(4)));

#ifdef __has_builtin
#  if __has_builtin(__builtin_amdgcn_cvt_pk_bf16_f32)
#    define HAVE_CVT_PK_BF16 1
#  endif
#endif
#ifndef HAVE_CVT_PK_BF16
#  define HAVE_CVT_PK_BF16 0
#endif

union BFVec {
    v16bf          v;
    unsigned short u[16];
    u32x4          q[2];
};

__device__ __forceinline__ unsigned short f2bf(float f) {
    unsigned int u = __float_as_uint(f);
    u += 0x7FFFu + ((u >> 16) & 1u);   // round-to-nearest-even
    return (unsigned short)(u >> 16);
}

// pack two f32 -> packed bf16x2 (v_cvt_pk_bf16_f32 when the builtin exists)
__device__ __forceinline__ unsigned int pack_bf16(float lo, float hi) {
#if HAVE_CVT_PK_BF16
    typedef __bf16 v2bf __attribute__((ext_vector_type(2)));
    v2bf p = __builtin_amdgcn_cvt_pk_bf16_f32(lo, hi);
    return __builtin_bit_cast(unsigned int, p);
#else
    return (unsigned)f2bf(lo) | ((unsigned)f2bf(hi) << 16);
#endif
}

// Load 16 bf16 (two 16B chunks) into a WMMA operand register set.
__device__ __forceinline__ v16bf ldpair(const unsigned short* p0, const unsigned short* p1) {
    BFVec t;
    t.q[0] = *(const u32x4*)p0;
    t.q[1] = *(const u32x4*)p1;
    return t.v;
}

__device__ __forceinline__ v8f wmma_bf16(v16bf a, v16bf b, v8f c) {
    return __builtin_amdgcn_wmma_f32_16x16x32_bf16(false, a, false, b, (short)0, c, false, false);
}

// async DMA: global -> LDS, 16 bytes per lane, tracked by ASYNCcnt
__device__ __forceinline__ void async_b128(unsigned lds, const unsigned short* g) {
    asm volatile("global_load_async_to_lds_b128 %0, %1, off" :: "v"(lds), "v"(g) : "memory");
}

// xor-shuffle within 16-lane halves via ds_swizzle (group-of-32 mode, and_mask=0x1F)
#define XORF(x, MASK) \
    __int_as_float(__builtin_amdgcn_ds_swizzle(__float_as_int(x), (0x1F | ((MASK) << 10))))

// ---------------------------------------------------------------------------
// Kernel 0: one-shot f32 -> bf16 conversion (8 elements / thread, b128 I/O).
// ---------------------------------------------------------------------------
__global__ __launch_bounds__(256) void cvt_bf16_kernel(
    const float* __restrict__ src, unsigned short* __restrict__ dst, int n8)
{
    const int i = blockIdx.x * blockDim.x + threadIdx.x;
    if (i >= n8) return;
    const f4* s = (const f4*)(src + (size_t)i * 8);
    f4 x0 = s[0];
    f4 x1 = s[1];
    u32x4 p = { pack_bf16(x0[0], x0[1]), pack_bf16(x0[2], x0[3]),
                pack_bf16(x1[0], x1[1]), pack_bf16(x1[2], x1[3]) };
    *(u32x4*)(dst + (size_t)i * 8) = p;
}

// ---------------------------------------------------------------------------
// Kernel 1: QKV projection GEMM, all-bf16 operands, f32 accum.
// grid = (8192/128, 1024/128, 3), block = 128 (4 waves as 2x2 quadrants).
// Each wave: 64x64 output (4x4 WMMA tiles, 16 accumulators), K swept by 32.
// ---------------------------------------------------------------------------
__global__ __launch_bounds__(128) void qkv_gemm_kernel(
    const unsigned short* __restrict__ hsb, const unsigned short* __restrict__ kvb,
    const unsigned short* __restrict__ Wqb, const unsigned short* __restrict__ Wkb,
    const unsigned short* __restrict__ Wvb,
    const float* __restrict__ bq, const float* __restrict__ bk,
    const float* __restrict__ bv,
    unsigned short* __restrict__ Qb, unsigned short* __restrict__ Kb,
    unsigned short* __restrict__ VTb)
{
    const int lane = threadIdx.x & 31;
    const int w    = threadIdx.x >> 5;
    const int m    = lane & 15;
    const int hi   = lane >> 4;
    const int wr   = w >> 1;
    const int wc   = w & 1;
    const int which = blockIdx.z;

    const unsigned short* X = (which == 0) ? hsb : kvb;
    const unsigned short* W = (which == 0) ? Wqb : (which == 1) ? Wkb : Wvb;
    const float* bias       = (which == 0) ? bq  : (which == 1) ? bk  : bv;

    const int row0 = blockIdx.x * 128 + wr * 64;   // rows over B*S = 8192
    const int o0   = blockIdx.y * 128 + wc * 64;   // output features

    const unsigned short* ar[4];
    const unsigned short* br[4];
#pragma unroll
    for (int i = 0; i < 4; ++i) {
        ar[i] = X + (size_t)(row0 + i * 16 + m) * H_;   // A rows, K-contiguous
        br[i] = W + (size_t)(o0 + i * 16 + m) * H_;     // B cols = W rows, K-contiguous
    }

    v8f acc[4][4];
#pragma unroll
    for (int i = 0; i < 4; ++i)
#pragma unroll
        for (int j = 0; j < 4; ++j) acc[i][j] = {};

    for (int k = 0; k < H_; k += 32) {
        v16bf bop[4];
#pragma unroll
        for (int ni = 0; ni < 4; ++ni) {
            // B layout: low lanes k=[0..15], high lanes k=[16..31] (contiguous)
            const unsigned short* p = br[ni] + k + hi * 16;
            bop[ni] = ldpair(p, p + 8);
        }
#pragma unroll
        for (int mi = 0; mi < 4; ++mi) {
            // A layout: low lanes k=[0..15], high lanes k=[16..31] (contiguous)
            const unsigned short* p = ar[mi] + k + hi * 16;
            const v16bf aop = ldpair(p, p + 8);
#pragma unroll
            for (int ni = 0; ni < 4; ++ni)
                acc[mi][ni] = wmma_bf16(aop, bop[ni], acc[mi][ni]);
        }
    }

    const int b = row0 >> 10;   // 128-row block never straddles a batch
#pragma unroll
    for (int ni = 0; ni < 4; ++ni) {
        const int o  = o0 + ni * 16 + m;
        const int nh = o >> 6;
        const int hd = o & 63;
        const float biasv = bias[o];
#pragma unroll
        for (int mi = 0; mi < 4; ++mi) {
            v8f a = acc[mi][ni];
#pragma unroll
            for (int r = 0; r < 8; ++r) a[r] += biasv;

            if (which == 2) {
                // V^T store: [b,h,hd,SKV]; rows r are contiguous seq -> one b128
                const int s0 = ((row0 + mi * 16) & 1023) + 8 * hi;
                u32x4 pk = { pack_bf16(a[0], a[1]), pack_bf16(a[2], a[3]),
                             pack_bf16(a[4], a[5]), pack_bf16(a[6], a[7]) };
                unsigned short* dst =
                    VTb + ((size_t)(b * NH_ + nh) * HD_ + hd) * SKV_ + s0;
                *(u32x4*)dst = pk;
            } else {
                unsigned short* Dst = (which == 0) ? Qb : Kb;
#pragma unroll
                for (int r = 0; r < 8; ++r) {
                    const int s = ((row0 + mi * 16) & 1023) + 8 * hi + r;
                    Dst[((size_t)(b * NH_ + nh) * SQ_ + s) * HD_ + hd] = f2bf(a[r]);
                }
            }
        }
    }
}

// ---------------------------------------------------------------------------
// Kernel 2: flash attention with async-DMA double-buffered K/V in LDS.
// grid = (SQ/64, NH, B), block = 128 (4 waves). Per 32-kv step: the block
// async-copies K(32x64) + V^T(64x32) tiles (8KB) into LDS, waves consume via
// ds_load_b128 -> WMMA. Double buffer overlaps DMA with compute.
// ---------------------------------------------------------------------------
__global__ __launch_bounds__(128) void attn_kernel(
    const unsigned short* __restrict__ Qb, const unsigned short* __restrict__ Kb,
    const unsigned short* __restrict__ VTb, const float* __restrict__ amask,
    float* __restrict__ out)
{
    __shared__ unsigned short Klds[2][32 * 64];    // 2 x 4KB, row-major [kv][d]
    __shared__ unsigned short Vlds[2][64 * 32];    // 2 x 4KB, row-major [d][kv]
    __shared__ unsigned short Plds[4 * 16 * 32];   // wave-private P tiles

    const int tid  = threadIdx.x;
    const int lane = tid & 31;
    const int w    = tid >> 5;
    const int m    = lane & 15;
    const int hi   = lane >> 4;
    const int b    = blockIdx.z;
    const int h    = blockIdx.y;
    const int q0   = blockIdx.x * 64 + w * 16;

    const unsigned short* Qh  = Qb  + (size_t)(b * NH_ + h) * SQ_  * HD_;
    const unsigned short* Kh  = Kb  + (size_t)(b * NH_ + h) * SKV_ * HD_;
    const unsigned short* VTh = VTb + (size_t)(b * NH_ + h) * HD_  * SKV_;
    const float* mrow = amask + (size_t)b * SKV_;

    // Q as A-operand, held for the whole KV sweep (d=0..31 and d=32..63)
    const unsigned short* qbase = Qh + (size_t)(q0 + m) * HD_;
    const v16bf qa0 = ldpair(qbase + hi * 16,      qbase + hi * 16 + 8);
    const v16bf qa1 = ldpair(qbase + 32 + hi * 16, qbase + 40 + hi * 16);

    // per-thread DMA slices: K tile is one contiguous 4KB block; V^T is 64x64B rows
    const int vrow = tid >> 1;            // 0..63 (d)
    const int vcol = (tid & 1) * 16;      // 0 / 16 (kv)

    auto issue_tile = [&](int kv, int buf) {
        const unsigned short* gk = Kh + (size_t)kv * HD_ + tid * 16;
        unsigned lk = (unsigned)(uintptr_t)&Klds[buf][tid * 16];
        async_b128(lk,       gk);
        async_b128(lk + 16u, gk + 8);
        const unsigned short* gv = VTh + (size_t)vrow * SKV_ + kv + vcol;
        unsigned lv = (unsigned)(uintptr_t)&Vlds[buf][vrow * 32 + vcol];
        async_b128(lv,       gv);
        async_b128(lv + 16u, gv + 8);
    };

    issue_tile(0, 0);

    v8f c[4] = {{}, {}, {}, {}};     // ctx f32: 16q x 64d (4 tiles of 16 cols)
    float rm[8], rl[8];
#pragma unroll
    for (int r = 0; r < 8; ++r) { rm[r] = -3.0e38f; rl[r] = 0.0f; }

    unsigned short* Pw = &Plds[w * 512];

    for (int kv = 0; kv < SKV_; kv += 32) {
        const int cur = (kv >> 5) & 1;
        // this wave's DMA for tile `cur` has landed in LDS
        asm volatile("s_wait_asynccnt 0x0" ::: "memory");
        // all waves' DMA done AND everyone finished consuming the other buffer
        __syncthreads();
        if (kv + 32 < SKV_) issue_tile(kv + 32, cur ^ 1);

        const unsigned short* kb = &Klds[cur][0];
        const unsigned short* vb = &Vlds[cur][0];

        // scores: two 16x16 tiles (cols kv..kv+15, kv+16..kv+31) from LDS
        const unsigned short* kp0 = kb + (size_t)m * 64;
        v8f s0 = {};
        s0 = wmma_bf16(qa0, ldpair(kp0 + hi * 16,      kp0 + hi * 16 + 8),      s0);
        s0 = wmma_bf16(qa1, ldpair(kp0 + 32 + hi * 16, kp0 + 32 + hi * 16 + 8), s0);
        const unsigned short* kp1 = kb + (size_t)(16 + m) * 64;
        v8f s1 = {};
        s1 = wmma_bf16(qa0, ldpair(kp1 + hi * 16,      kp1 + hi * 16 + 8),      s1);
        s1 = wmma_bf16(qa1, ldpair(kp1 + 32 + hi * 16, kp1 + 32 + hi * 16 + 8), s1);

        const float mk0 = mrow[kv + m];
        const float mk1 = mrow[kv + 16 + m];
#pragma unroll
        for (int r = 0; r < 8; ++r) {
            s0[r] = s0[r] * 0.125f + mk0;   // 1/sqrt(64)
            s1[r] = s1[r] * 0.125f + mk1;
        }

        // online softmax: row reductions across the 16 lanes of each half-wave
#pragma unroll
        for (int r = 0; r < 8; ++r) {
            float t = fmaxf(s0[r], s1[r]);
            t = fmaxf(t, XORF(t, 1)); t = fmaxf(t, XORF(t, 2));
            t = fmaxf(t, XORF(t, 4)); t = fmaxf(t, XORF(t, 8));
            const float nm   = fmaxf(rm[r], t);
            const float corr = __expf(rm[r] - nm);
            rm[r] = nm;
            const float p0 = __expf(s0[r] - nm);
            const float p1 = __expf(s1[r] - nm);
            s0[r] = p0; s1[r] = p1;
            float ps = p0 + p1;
            ps += XORF(ps, 1); ps += XORF(ps, 2); ps += XORF(ps, 4); ps += XORF(ps, 8);
            rl[r] = rl[r] * corr + ps;
            c[0][r] *= corr; c[1][r] *= corr; c[2][r] *= corr; c[3][r] *= corr;
        }

        // C-layout -> A-layout for P via wave-private LDS (16 rows x 32 cols bf16)
#pragma unroll
        for (int r = 0; r < 8; ++r) {
            Pw[(8 * hi + r) * 32 + m]      = f2bf(s0[r]);
            Pw[(8 * hi + r) * 32 + 16 + m] = f2bf(s1[r]);
        }
        asm volatile("s_wait_dscnt 0x0" ::: "memory");
        const unsigned short* pp = Pw + m * 32;
        const v16bf pa = ldpair(pp + hi * 16, pp + hi * 16 + 8);

        // ctx += P @ V from LDS V^T (rows kv-contiguous per lane)
#pragma unroll
        for (int dt = 0; dt < 4; ++dt) {
            const unsigned short* vp = vb + (size_t)(dt * 16 + m) * 32 + hi * 16;
            c[dt] = wmma_bf16(pa, ldpair(vp, vp + 8), c[dt]);
        }
    }

    // epilogue: ctx / l, store fp32 out [B, SQ, NH*HD]
    float inv[8];
#pragma unroll
    for (int r = 0; r < 8; ++r) inv[r] = 1.0f / rl[r];
#pragma unroll
    for (int dt = 0; dt < 4; ++dt) {
#pragma unroll
        for (int r = 0; r < 8; ++r) {
            const int qrow = q0 + 8 * hi + r;
            out[((size_t)(b * SQ_ + qrow)) * H_ + h * HD_ + dt * 16 + m] = c[dt][r] * inv[r];
        }
    }
}

// ---------------------------------------------------------------------------
extern "C" void kernel_launch(void* const* d_in, const int* in_sizes, int n_in,
                              void* d_out, int out_size, void* d_ws, size_t ws_size,
                              hipStream_t stream) {
    const float* hs   = (const float*)d_in[0];
    const float* kvs  = (const float*)d_in[1];
    const float* mask = (const float*)d_in[2];
    const float* Wq   = (const float*)d_in[3];
    const float* bq   = (const float*)d_in[4];
    const float* Wk   = (const float*)d_in[5];
    const float* bk   = (const float*)d_in[6];
    const float* Wv   = (const float*)d_in[7];
    const float* bv   = (const float*)d_in[8];
    float* out        = (float*)d_out;

    const size_t nAct = (size_t)B_ * SQ_ * H_;        // 8,388,608 (hs and kv each)
    const size_t nW   = (size_t)H_ * H_;              // 1,048,576 per weight
    const size_t nQKV = (size_t)B_ * NH_ * SQ_ * HD_; // 8,388,608 per Q/K/VT

    unsigned short* Qb  = (unsigned short*)d_ws;
    unsigned short* Kb  = Qb  + nQKV;
    unsigned short* VTb = Kb  + nQKV;
    unsigned short* hsb = VTb + nQKV;
    unsigned short* kvb = hsb + nAct;
    unsigned short* Wqb = kvb + nAct;
    unsigned short* Wkb = Wqb + nW;
    unsigned short* Wvb = Wkb + nW;
    const size_t need = (3 * nQKV + 2 * nAct + 3 * nW) * sizeof(unsigned short);
    if (ws_size < need) return;

    // one-shot bf16 conversion of activations and weights
    cvt_bf16_kernel<<<dim3((unsigned)(nAct / 8 / 256)), dim3(256), 0, stream>>>(hs,  hsb, (int)(nAct / 8));
    cvt_bf16_kernel<<<dim3((unsigned)(nAct / 8 / 256)), dim3(256), 0, stream>>>(kvs, kvb, (int)(nAct / 8));
    cvt_bf16_kernel<<<dim3((unsigned)(nW / 8 / 256)),   dim3(256), 0, stream>>>(Wq,  Wqb, (int)(nW / 8));
    cvt_bf16_kernel<<<dim3((unsigned)(nW / 8 / 256)),   dim3(256), 0, stream>>>(Wk,  Wkb, (int)(nW / 8));
    cvt_bf16_kernel<<<dim3((unsigned)(nW / 8 / 256)),   dim3(256), 0, stream>>>(Wv,  Wvb, (int)(nW / 8));

    dim3 g1((B_ * SQ_) / 128, H_ / 128, 3);
    qkv_gemm_kernel<<<g1, dim3(128), 0, stream>>>(hsb, kvb, Wqb, Wkb, Wvb,
                                                  bq, bk, bv, Qb, Kb, VTb);

    dim3 g2(SQ_ / 64, NH_, B_);
    attn_kernel<<<g2, dim3(128), 0, stream>>>(Qb, Kb, VTb, mask, out);
}
